// DINAttention_31619549233560
// MI455X (gfx1250) — matrix operand order repaired and checked by reading, not compiled, not run on hardware
//
#include <hip/hip_runtime.h>
#include <math.h>

typedef __attribute__((ext_vector_type(16))) __bf16 v16bf;
typedef __attribute__((ext_vector_type(8)))  float  v8f;

union AFrag { __bf16 h[16]; v16bf v; };

#define NBT 409600            // B*T = 2048*200
#define INV_NBT (1.0f/409600.0f)
#define PADV (-4294967295.0f)
#define G1_GRID 800
#define G1_TILES 3200
#define G1_LDS_BYTES 148096   // 40960(w1hi)+40960(w1lo)+65536(q/k)+640(stats)

static __device__ __forceinline__ float diceAct(float x, float mean, float rstd, float alpha) {
    float xn = (x - mean) * rstd;
    float p  = 1.0f / (1.0f + __expf(-xn));
    return (p + alpha * (1.0f - p)) * x;
}

// split-bf16 helpers: x ~= hi + lo with hi,lo representable in bf16
static __device__ __forceinline__ void splitbf(float x, __bf16& h, __bf16& l) {
    h = (__bf16)x;
    l = (__bf16)(x - (float)h);
}

// ---------------------------------------------------------------------------
// Kernel 0: zero the stat accumulators (sum1[80], sq1[80], sum2[48], sq2[48])
// ---------------------------------------------------------------------------
__global__ void init_stats_kernel(float* __restrict__ ws) {
    int i = threadIdx.x;
    if (i < 256) ws[i] = 0.0f;
}

// ---------------------------------------------------------------------------
// Kernel 1: H1 = X @ W1 + b1, X = [q,k,q-k,q*k] built on the fly.
// M=409600, K=256, N=80. Split-bf16 WMMA (AhBh + AlBh + AhBl), f32 accum.
// Persistent blocks (grid-stride over 128-row tiles); per-wave q/k rows are
// staged to LDS with global_load_async_to_lds_b128 (ASYNCcnt path).
// ---------------------------------------------------------------------------
__global__ __launch_bounds__(256) void gemm1_kernel(
    const float* __restrict__ qmat,   // [2048,64]
    const float* __restrict__ keys,   // [2048,200,64]
    const float* __restrict__ W1,     // [256,80]
    const float* __restrict__ b1,     // [80]
    float* __restrict__ h1,           // [409600,80]
    float* __restrict__ gsum, float* __restrict__ gsq)
{
    extern __shared__ char smem[];
    __bf16* w1hi = (__bf16*)(smem);            // [128*160] kp-pair x (2n+j)
    __bf16* w1lo = (__bf16*)(smem + 40960);    // [128*160]
    float*  sSum = (float*)(smem + 147456);    // [80]
    float*  sSq  = sSum + 80;                  // [80]

    const int tid = threadIdx.x;
    // Stage W1 into LDS as split-bf16 K-pairs (B-fragment layout)
    for (int idx = tid; idx < 128 * 80; idx += 256) {
        int kp = idx / 80, n = idx % 80;
        float w0 = W1[(2 * kp + 0) * 80 + n];
        float w1v = W1[(2 * kp + 1) * 80 + n];
        __bf16 h0, l0, h1v, l1;
        splitbf(w0, h0, l0);
        splitbf(w1v, h1v, l1);
        w1hi[kp * 160 + 2 * n + 0] = h0;
        w1hi[kp * 160 + 2 * n + 1] = h1v;
        w1lo[kp * 160 + 2 * n + 0] = l0;
        w1lo[kp * 160 + 2 * n + 1] = l1;
    }
    if (tid < 80) { sSum[tid] = 0.0f; sSq[tid] = 0.0f; }
    __syncthreads();

    const int wave = tid >> 5;
    const int lane = tid & 31;
    const int hi   = (lane >= 16) ? 8 : 0;
    const unsigned qByte = (unsigned)(size_t)smem + 81920u + (unsigned)wave * 8192u;
    const float* qrow = (const float*)(smem + 81920 + wave * 8192) + (lane & 15) * 64;
    const float* krow = qrow + 1024;

    float bias[5], sumR[5], sqR[5];
    #pragma unroll
    for (int nt = 0; nt < 5; ++nt) {
        bias[nt] = b1[nt * 16 + (lane & 15)];
        sumR[nt] = 0.0f; sqR[nt] = 0.0f;
    }
    const v8f vzero = {0.f,0.f,0.f,0.f,0.f,0.f,0.f,0.f};

    for (int tt = blockIdx.x; tt < G1_TILES; tt += gridDim.x) {
        const int rowBase = tt * 128 + wave * 16;

        // Previous tile's LDS reads must complete before overwriting staging
        asm volatile("s_wait_dscnt 0x0" ::: "memory");
        // Async-stage this wave's 16 q-rows and 16 k-rows (64 floats each)
        #pragma unroll
        for (int i = 0; i < 8; ++i) {
            int chunk = i * 32 + lane;            // 0..255 : row = chunk/16, 4-float col chunk
            int row = chunk >> 4;
            int c4  = (chunk & 15) * 4;
            int m   = rowBase + row;
            int bb  = m / 200, t = m % 200;
            const float* gq = qmat + (size_t)bb * 64 + c4;
            const float* gk = keys + ((size_t)bb * 200 + t) * 64 + c4;
            unsigned lq = qByte + (unsigned)chunk * 16u;
            unsigned lk = lq + 4096u;
            asm volatile("global_load_async_to_lds_b128 %0, %1, off"
                         :: "v"(lq), "v"((unsigned long long)(size_t)gq) : "memory");
            asm volatile("global_load_async_to_lds_b128 %0, %1, off"
                         :: "v"(lk), "v"((unsigned long long)(size_t)gk) : "memory");
        }
        asm volatile("s_wait_asynccnt 0x0" ::: "memory");

        v8f acc[5];
        #pragma unroll
        for (int nt = 0; nt < 5; ++nt) acc[nt] = vzero;

        #pragma unroll
        for (int s = 0; s < 8; ++s) {
            const int blk = s >> 1;        // 0:q 1:k 2:q-k 3:q*k
            const int off = (s & 1) * 32;
            const int klo = off + 2 * hi;
            const int khi = off + 2 * hi + 8;
            AFrag aH, aL;
            #pragma unroll
            for (int i = 0; i < 8; ++i) {
                float qv = qrow[klo + i], kv = krow[klo + i];
                float x  = (blk == 0) ? qv : (blk == 1) ? kv : (blk == 2) ? (qv - kv) : (qv * kv);
                splitbf(x, aH.h[i], aL.h[i]);
                float qv2 = qrow[khi + i], kv2 = krow[khi + i];
                float x2  = (blk == 0) ? qv2 : (blk == 1) ? kv2 : (blk == 2) ? (qv2 - kv2) : (qv2 * kv2);
                splitbf(x2, aH.h[8 + i], aL.h[8 + i]);
            }
            #pragma unroll
            for (int nt = 0; nt < 5; ++nt) {
                AFrag bH, bL;
                const int n = nt * 16 + (lane & 15);
                #pragma unroll
                for (int v = 0; v < 8; ++v) {
                    const int kp = s * 16 + hi + v;
                    bH.h[2 * v]     = w1hi[kp * 160 + 2 * n];
                    bH.h[2 * v + 1] = w1hi[kp * 160 + 2 * n + 1];
                    bL.h[2 * v]     = w1lo[kp * 160 + 2 * n];
                    bL.h[2 * v + 1] = w1lo[kp * 160 + 2 * n + 1];
                }
                acc[nt] = __builtin_amdgcn_wmma_f32_16x16x32_bf16(
                    false, aH.v, false, bH.v, (short)0, acc[nt], false, false);
                acc[nt] = __builtin_amdgcn_wmma_f32_16x16x32_bf16(
                    false, aL.v, false, bH.v, (short)0, acc[nt], false, false);
                acc[nt] = __builtin_amdgcn_wmma_f32_16x16x32_bf16(
                    false, aH.v, false, bL.v, (short)0, acc[nt], false, false);
            }
        }

        // Store tile + accumulate per-feature stats in registers
        #pragma unroll
        for (int nt = 0; nt < 5; ++nt) {
            const int n = nt * 16 + (lane & 15);
            #pragma unroll
            for (int r = 0; r < 8; ++r) {
                const int m = rowBase + hi + r;   // lanes16-31 -> M = 8+r
                float v = acc[nt][r] + bias[nt];
                h1[(size_t)m * 80 + n] = v;
                sumR[nt] += v;
                sqR[nt]  += v * v;
            }
        }
    }

    // Block-level stat reduction: one LDS atomic per accumulator, then global
    #pragma unroll
    for (int nt = 0; nt < 5; ++nt) {
        const int n = nt * 16 + (lane & 15);
        atomicAdd(&sSum[n], sumR[nt]);
        atomicAdd(&sSq[n],  sqR[nt]);
    }
    __syncthreads();
    if (tid < 80) { atomicAdd(&gsum[tid], sSum[tid]); atomicAdd(&gsq[tid], sSq[tid]); }
}

// ---------------------------------------------------------------------------
// Kernel 2: H2 = dice(H1) @ W2 + b2.  K=80 (pad 96), N=40 (pad 48).
// Split-bf16 WMMA; dice layer-1 applied while building A fragments.
// ---------------------------------------------------------------------------
__global__ __launch_bounds__(256) void gemm2_kernel(
    const float* __restrict__ h1,
    const float* __restrict__ gsum1, const float* __restrict__ gsq1,
    const float* __restrict__ a1,
    const float* __restrict__ W2,    // [80,40]
    const float* __restrict__ b2,    // [40]
    float* __restrict__ h2,          // [409600,48] (cols 40..47 unused)
    float* __restrict__ gsum2, float* __restrict__ gsq2)
{
    __shared__ __bf16 w2hi[48 * 96];
    __shared__ __bf16 w2lo[48 * 96];
    __shared__ float meanS[80], rstdS[80], alphaS[80];
    __shared__ float sSum[48], sSq[48];

    const int tid = threadIdx.x;
    for (int idx = tid; idx < 48 * 48; idx += 256) {
        int kp = idx / 48, n = idx % 48;
        int k0 = 2 * kp, k1 = 2 * kp + 1;
        float w0 = (k0 < 80 && n < 40) ? W2[k0 * 40 + n] : 0.0f;
        float w1 = (k1 < 80 && n < 40) ? W2[k1 * 40 + n] : 0.0f;
        __bf16 h0, l0, h1v, l1;
        splitbf(w0, h0, l0);
        splitbf(w1, h1v, l1);
        w2hi[kp * 96 + 2 * n + 0] = h0;
        w2hi[kp * 96 + 2 * n + 1] = h1v;
        w2lo[kp * 96 + 2 * n + 0] = l0;
        w2lo[kp * 96 + 2 * n + 1] = l1;
    }
    if (tid < 80) {
        float s = gsum1[tid], q = gsq1[tid];
        float mean = s * INV_NBT;
        float var  = q * INV_NBT - mean * mean;
        meanS[tid] = mean;
        rstdS[tid] = rsqrtf(var + 1e-8f);
        alphaS[tid] = a1[tid];
    }
    if (tid < 48) { sSum[tid] = 0.0f; sSq[tid] = 0.0f; }
    __syncthreads();

    const int wave = tid >> 5;
    const int lane = tid & 31;
    const int hi   = (lane >= 16) ? 8 : 0;
    const v8f vzero = {0.f,0.f,0.f,0.f,0.f,0.f,0.f,0.f};

    float bias[3], sumR[3], sqR[3];
    #pragma unroll
    for (int nt = 0; nt < 3; ++nt) {
        const int n = nt * 16 + (lane & 15);
        bias[nt] = (n < 40) ? b2[n] : 0.0f;
        sumR[nt] = 0.0f; sqR[nt] = 0.0f;
    }

    for (int tt = blockIdx.x; tt < G1_TILES; tt += gridDim.x) {
        const int rowBase = tt * 128 + wave * 16;
        const float* hr = h1 + (size_t)(rowBase + (lane & 15)) * 80;

        v8f acc[3];
        #pragma unroll
        for (int nt = 0; nt < 3; ++nt) acc[nt] = vzero;

        #pragma unroll
        for (int s = 0; s < 3; ++s) {
            const int klo = s * 32 + 2 * hi;
            const int khi = s * 32 + 2 * hi + 8;
            AFrag aH, aL;
            #pragma unroll
            for (int i = 0; i < 8; ++i) {
                int f = klo + i;
                float x0 = (f < 80) ? diceAct(hr[f], meanS[f], rstdS[f], alphaS[f]) : 0.0f;
                splitbf(x0, aH.h[i], aL.h[i]);
                int f2 = khi + i;
                float x1 = (f2 < 80) ? diceAct(hr[f2], meanS[f2], rstdS[f2], alphaS[f2]) : 0.0f;
                splitbf(x1, aH.h[8 + i], aL.h[8 + i]);
            }
            #pragma unroll
            for (int nt = 0; nt < 3; ++nt) {
                AFrag bH, bL;
                const int n = nt * 16 + (lane & 15);
                #pragma unroll
                for (int v = 0; v < 8; ++v) {
                    const int kp = s * 16 + hi + v;
                    bH.h[2 * v]     = w2hi[kp * 96 + 2 * n];
                    bH.h[2 * v + 1] = w2hi[kp * 96 + 2 * n + 1];
                    bL.h[2 * v]     = w2lo[kp * 96 + 2 * n];
                    bL.h[2 * v + 1] = w2lo[kp * 96 + 2 * n + 1];
                }
                acc[nt] = __builtin_amdgcn_wmma_f32_16x16x32_bf16(
                    false, aH.v, false, bH.v, (short)0, acc[nt], false, false);
                acc[nt] = __builtin_amdgcn_wmma_f32_16x16x32_bf16(
                    false, aL.v, false, bH.v, (short)0, acc[nt], false, false);
                acc[nt] = __builtin_amdgcn_wmma_f32_16x16x32_bf16(
                    false, aH.v, false, bL.v, (short)0, acc[nt], false, false);
            }
        }

        #pragma unroll
        for (int nt = 0; nt < 3; ++nt) {
            const int n = nt * 16 + (lane & 15);
            if (n < 40) {
                #pragma unroll
                for (int r = 0; r < 8; ++r) {
                    const int m = rowBase + hi + r;
                    float v = acc[nt][r] + bias[nt];
                    h2[(size_t)m * 48 + n] = v;
                    sumR[nt] += v;
                    sqR[nt]  += v * v;
                }
            }
        }
    }

    #pragma unroll
    for (int nt = 0; nt < 3; ++nt) {
        const int n = nt * 16 + (lane & 15);
        if (n < 40) {
            atomicAdd(&sSum[n], sumR[nt]);
            atomicAdd(&sSq[n],  sqR[nt]);
        }
    }
    __syncthreads();
    if (tid < 40) { atomicAdd(&gsum2[tid], sSum[tid]); atomicAdd(&gsq2[tid], sSq[tid]); }
}

// ---------------------------------------------------------------------------
// Kernel 3: per batch row — dice(H2)@W3 + b3, mask, softmax over T=200,
// out[b,:] = sum_t keys[b,t,:] * w[t].
// ---------------------------------------------------------------------------
__global__ __launch_bounds__(256) void attn_kernel(
    const float* __restrict__ keys,
    const unsigned char* __restrict__ mask,
    const float* __restrict__ h2,
    const float* __restrict__ gsum2, const float* __restrict__ gsq2,
    const float* __restrict__ a2,
    const float* __restrict__ W3, const float* __restrict__ b3,
    float* __restrict__ out)
{
    __shared__ float mean2[40], rstd2[40], alpha2[40], w3s[40];
    __shared__ float sc[256];
    __shared__ float red[256];
    __shared__ float part[4][64];

    const int tid = threadIdx.x;
    const int b = blockIdx.x;

    if (tid < 40) {
        float s = gsum2[tid], q = gsq2[tid];
        float mean = s * INV_NBT;
        float var  = q * INV_NBT - mean * mean;
        mean2[tid] = mean;
        rstd2[tid] = rsqrtf(var + 1e-8f);
        alpha2[tid] = a2[tid];
        w3s[tid] = W3[tid];
    }
    __syncthreads();

    float score = -INFINITY;
    if (tid < 200) {
        const float* hr = h2 + ((size_t)b * 200 + tid) * 48;
        float acc = b3[0];
        #pragma unroll 8
        for (int f = 0; f < 40; ++f)
            acc += diceAct(hr[f], mean2[f], rstd2[f], alpha2[f]) * w3s[f];
        score = mask[b * 200 + tid] ? acc : PADV;
    }
    sc[tid] = score;
    red[tid] = score;
    __syncthreads();
    for (int off = 128; off > 0; off >>= 1) {
        if (tid < off) red[tid] = fmaxf(red[tid], red[tid + off]);
        __syncthreads();
    }
    const float mx = red[0];
    __syncthreads();
    float e = (tid < 200) ? __expf(sc[tid] - mx) : 0.0f;
    red[tid] = e;
    __syncthreads();
    for (int off = 128; off > 0; off >>= 1) {
        if (tid < off) red[tid] += red[tid + off];
        __syncthreads();
    }
    const float denom = red[0];
    __syncthreads();
    sc[tid] = e / denom;            // normalized attention weight
    __syncthreads();

    const int d = tid & 63;
    const int g = tid >> 6;         // 4 t-chunks of 50
    float accO = 0.0f;
    for (int t = g * 50; t < (g + 1) * 50; ++t)
        accO += keys[((size_t)b * 200 + t) * 64 + d] * sc[t];
    part[g][d] = accO;
    __syncthreads();
    if (tid < 64)
        out[(size_t)b * 64 + tid] = part[0][tid] + part[1][tid] + part[2][tid] + part[3][tid];
}

// ---------------------------------------------------------------------------
extern "C" void kernel_launch(void* const* d_in, const int* in_sizes, int n_in,
                              void* d_out, int out_size, void* d_ws, size_t ws_size,
                              hipStream_t stream)
{
    const float* q    = (const float*)d_in[0];
    const float* keys = (const float*)d_in[1];
    const unsigned char* mask = (const unsigned char*)d_in[2];
    const float* W1 = (const float*)d_in[3];
    const float* b1 = (const float*)d_in[4];
    const float* a1 = (const float*)d_in[5];
    const float* W2 = (const float*)d_in[6];
    const float* b2 = (const float*)d_in[7];
    const float* a2 = (const float*)d_in[8];
    const float* W3 = (const float*)d_in[9];
    const float* b3 = (const float*)d_in[10];

    float* ws   = (float*)d_ws;
    float* sum1 = ws;          // [80]
    float* sq1  = ws + 80;     // [80]
    float* sum2 = ws + 160;    // [48]
    float* sq2  = ws + 208;    // [48]
    float* h1   = ws + 256;                         // [409600*80]
    float* h2   = h1 + (size_t)NBT * 80;            // [409600*48]
    float* out  = (float*)d_out;

    // Opt in to >64KB dynamic LDS for the staged GEMM1 kernel (CDNA5: 320KB/WGP)
    (void)hipFuncSetAttribute((const void*)gemm1_kernel,
                              hipFuncAttributeMaxDynamicSharedMemorySize,
                              G1_LDS_BYTES);

    init_stats_kernel<<<1, 256, 0, stream>>>(ws);
    gemm1_kernel<<<G1_GRID, 256, G1_LDS_BYTES, stream>>>(q, keys, W1, b1, h1, sum1, sq1);
    gemm2_kernel<<<G1_GRID, 256, 0, stream>>>(h1, sum1, sq1, a1, W2, b2, h2, sum2, sq2);
    attn_kernel<<<2048, 256, 0, stream>>>(keys, mask, h2, sum2, sq2, a2, W3, b3, out);
}
